// AttentionEdgePredictionHead_78314433675288
// MI455X (gfx1250) — matrix operand order repaired and checked by reading, not compiled, not run on hardware
//
#include <hip/hip_runtime.h>
#include <hip/hip_bf16.h>

// ---------------------------------------------------------------------------
// AttentionEdgePredictionHead for MI455X (gfx1250, wave32, WMMA bf16)
//
// Key algebra: pairs@W1 = node@W1top (per-b) + msg@W1bot (per-f); training-mode
// BN cancels b1/b2 and BN1 stats factor into column stats of A and B. The only
// large GEMM left is h1[131072,512] @ W2[512,256], done with
// v_wmma_f32_16x16x32_bf16, h1 generated on the fly from A[b]+B''[f].
// ---------------------------------------------------------------------------

typedef __attribute__((ext_vector_type(16))) __bf16 v16bf;
typedef __attribute__((ext_vector_type(8)))  float  v8f;

#define BSZ   2048
#define NFEAT 64
#define DDIM  512
#define EEDGE 1024
#define MTOT  (BSZ * NFEAT)      // 131072
#define EPSV  1e-5f

union FragU { unsigned u[8]; v16bf v; };

__device__ __forceinline__ unsigned short f2bf(float f) {
  unsigned u = __float_as_uint(f);
  u += 0x7FFFu + ((u >> 16) & 1u);          // round-to-nearest-even
  return (unsigned short)(u >> 16);
}
__device__ __forceinline__ unsigned f2ord(float f) {   // order-preserving f32->u32
  unsigned u = __float_as_uint(f);
  return (u & 0x80000000u) ? ~u : (u | 0x80000000u);
}
__device__ __forceinline__ float ord2f(unsigned k) {
  unsigned u = (k & 0x80000000u) ? (k & 0x7FFFFFFFu) : ~k;
  return __uint_as_float(u);
}
__device__ __forceinline__ float lrelu(float x, float s) { return x > 0.f ? x : s * x; }
// 16-bit A-matrix 16x32 layout: lane-half 0 -> K 0..7 / 16..23, half 1 -> 8..15 / 24..31
__device__ __forceinline__ int akidx0(int v, int half) {
  return (v < 4) ? (2 * v + 8 * half) : (16 + 2 * (v - 4) + 8 * half);
}

// ---------------------------------------------------------------- utility ---
__global__ void k_zero(float* p, int n) {
  int i = blockIdx.x * 256 + threadIdx.x;
  if (i < n) p[i] = 0.f;
}

__global__ void k_cvt_bf(const float* __restrict__ x, unsigned short* __restrict__ y, int n) {
  int i = blockIdx.x * 256 + threadIdx.x;
  if (i < n) y[i] = f2bf(x[i]);
}

// Pre-swizzle a K=512 x N weight (row-major f32) into WMMA B-fragment order:
// out[((kb*NT + nt)*32 + lane)*8 + v] = pack(bf16 W[k0,n], bf16 W[k0+1,n])
// with k0 = kb*32 + (lane>>4)*8 + 2v + (v>=4 ? 8 : 0), n = nt*16 + (lane&15).
__global__ void k_swz(const float* __restrict__ W, unsigned* __restrict__ out, int NTtot) {
  int i = blockIdx.x * 256 + threadIdx.x;
  int N = NTtot * 16;
  int v = i & 7, lane = (i >> 3) & 31, rem = i >> 8;
  int nt = rem % NTtot, kb = rem / NTtot;
  int k0 = kb * 32 + ((lane >> 4) << 3) + 2 * v + ((v >= 4) ? 8 : 0);
  int n  = nt * 16 + (lane & 15);
  unsigned lo = f2bf(W[(size_t)k0 * N + n]);
  unsigned hi = f2bf(W[(size_t)(k0 + 1) * N + n]);
  out[i] = lo | (hi << 16);
}

// ------------------------------------------------------------------- GAT ----
// h = feature_emb @ W_gat   (64 x 512) -- one block per feature row
__global__ void k_gat_lin(const float* __restrict__ feat, const float* __restrict__ Wg,
                          float* __restrict__ hfeat) {
  __shared__ float xrow[512];
  int n = blockIdx.x, tid = threadIdx.x;
  for (int k = tid; k < 512; k += 256) xrow[k] = feat[(size_t)n * 512 + k];
  __syncthreads();
  for (int c = tid; c < 512; c += 256) {
    float acc = 0.f;
    for (int k = 0; k < 512; ++k) acc += xrow[k] * Wg[(size_t)k * 512 + c];
    hfeat[(size_t)n * 512 + c] = acc;
  }
}

// a_src/a_dst per (feature, head) -- 1 block, 128 threads
__global__ void k_gat_att(const float* __restrict__ hfeat, const float* __restrict__ atts,
                          const float* __restrict__ attd, float* asrc, float* adst) {
  int t = threadIdx.x;
  if (t >= 128) return;
  int n = t >> 1, hh = t & 1;
  float s1 = 0.f, s2 = 0.f;
  for (int c = 0; c < 256; ++c) {
    float hv = hfeat[(size_t)n * 512 + hh * 256 + c];
    s1 += hv * atts[hh * 256 + c];
    s2 += hv * attd[hh * 256 + c];
  }
  asrc[n * 2 + hh] = s1;
  adst[n * 2 + hh] = s2;
}

// edge-wise scatter-softmax weights -- single block, segment max/sum in LDS
__global__ void k_gat_soft(const int* __restrict__ ri, const float* __restrict__ asrc,
                           const float* __restrict__ adst, float* __restrict__ wedge) {
  __shared__ unsigned ukey[128];
  __shared__ float denom[128];
  int tid = threadIdx.x;
  if (tid < 128) { ukey[tid] = 0u; denom[tid] = 0.f; }
  __syncthreads();
  for (int e = tid; e < EEDGE; e += 256) {
    int s = ri[e], d = ri[EEDGE + e];
    for (int hh = 0; hh < 2; ++hh) {
      float al = lrelu(asrc[s * 2 + hh] + adst[d * 2 + hh], 0.2f);
      wedge[e * 2 + hh] = al;
      atomicMax(&ukey[d * 2 + hh], f2ord(al));
    }
  }
  __syncthreads();
  for (int e = tid; e < EEDGE; e += 256) {
    int d = ri[EEDGE + e];
    for (int hh = 0; hh < 2; ++hh) {
      float ex = __expf(wedge[e * 2 + hh] - ord2f(ukey[d * 2 + hh]));
      wedge[e * 2 + hh] = ex;
      atomicAdd(&denom[d * 2 + hh], ex);
    }
  }
  __syncthreads();
  for (int e = tid; e < EEDGE; e += 256) {
    int d = ri[EEDGE + e];
    for (int hh = 0; hh < 2; ++hh) wedge[e * 2 + hh] /= denom[d * 2 + hh];
  }
}

// weighted scatter-add of messages -- one block per edge
__global__ void k_gat_agg(const int* __restrict__ ri, const float* __restrict__ hfeat,
                          const float* __restrict__ wedge, float* __restrict__ msg) {
  int e = blockIdx.x, c = threadIdx.x;
  int s = ri[e], d = ri[EEDGE + e];
  for (int hh = 0; hh < 2; ++hh)
    atomicAdd(&msg[(size_t)d * 512 + hh * 256 + c],
              hfeat[(size_t)s * 512 + hh * 256 + c] * wedge[e * 2 + hh]);
}

// msg + b_gat -> bf16 for the W1bot GEMM
__global__ void k_msgfinal(const float* __restrict__ msg, const float* __restrict__ bgat,
                           unsigned short* __restrict__ msg_bf) {
  int i = blockIdx.x * 256 + threadIdx.x;   // 32768
  int k = i & 511;
  msg_bf[i] = f2bf(msg[i] + bgat[k]);
}

// ------------------------------------------------- generic WMMA GEMM --------
// Out[M, 16*NTtot] = Xbf[M,512] @ W (pre-swizzled).  grid = (M/64, NTtot/16).
// 8 waves/block: wave w -> M-tile (w>>1), 8 N-tiles in half (w&1).
__global__ void k_gemm(const unsigned short* __restrict__ Xbf,
                       const unsigned* __restrict__ Wswz,
                       float* __restrict__ Out, int NTtot) {
  const int Nout = NTtot * 16;
  const int tid = threadIdx.x, lane = tid & 31, w = tid >> 5;
  const int mt = w >> 1, nh = w & 1;
  const int half = lane >> 4, mloc = lane & 15;
  const int row0 = blockIdx.x * 64;
  const int arow = row0 + mt * 16 + mloc;
  v8f zero = {0.f, 0.f, 0.f, 0.f, 0.f, 0.f, 0.f, 0.f};
  v8f acc[8];
#pragma unroll
  for (int i = 0; i < 8; ++i) acc[i] = zero;
  for (int kb = 0; kb < 16; ++kb) {
    FragU a;
#pragma unroll
    for (int v = 0; v < 8; ++v) {
      int k0 = kb * 32 + akidx0(v, half);
      a.u[v] = *(const unsigned*)(Xbf + (size_t)arow * 512 + k0);
    }
#pragma unroll
    for (int i = 0; i < 8; ++i) {
      int nt = blockIdx.y * 16 + nh * 8 + i;
      const uint4* p = (const uint4*)(Wswz + (((size_t)kb * NTtot + nt) * 32 + lane) * 8);
      uint4 q0 = p[0], q1 = p[1];
      FragU b;
      b.u[0] = q0.x; b.u[1] = q0.y; b.u[2] = q0.z; b.u[3] = q0.w;
      b.u[4] = q1.x; b.u[5] = q1.y; b.u[6] = q1.z; b.u[7] = q1.w;
      acc[i] = __builtin_amdgcn_wmma_f32_16x16x32_bf16(false, a.v, false, b.v,
                                                       (short)0, acc[i], false, false);
    }
  }
#pragma unroll
  for (int i = 0; i < 8; ++i) {
    int col = blockIdx.y * 256 + (nh * 8 + i) * 16 + mloc;
#pragma unroll
    for (int r = 0; r < 8; ++r) {
      int m = r + 8 * half;
      Out[(size_t)(row0 + mt * 16 + m) * Nout + col] = acc[i][r];
    }
  }
}

// ----------------------------------------------------- BN1 closed form ------
// Column sums / sumsq of A (2048 rows) and Brow (64 rows).
__global__ void k_stats1(const float* __restrict__ Amat, const float* __restrict__ Brow,
                         float* sumA, float* sqA, float* sumB, float* sqB) {
  __shared__ float rs[256], rq[256];
  const int tid = threadIdx.x;
  const bool isA = blockIdx.x < 512;
  const int k = isA ? blockIdx.x : blockIdx.x - 512;
  const float* M = isA ? Amat : Brow;
  const int rows = isA ? BSZ : NFEAT;
  float s = 0.f, q = 0.f;
  for (int r = tid; r < rows; r += 256) {
    float v = M[(size_t)r * 512 + k];
    s += v; q += v * v;
  }
  rs[tid] = s; rq[tid] = q;
  __syncthreads();
  for (int o = 128; o > 0; o >>= 1) {
    if (tid < o) { rs[tid] += rs[tid + o]; rq[tid] += rq[tid + o]; }
    __syncthreads();
  }
  if (tid == 0) {
    if (isA) { sumA[k] = rs[0]; sqA[k] = rq[0]; }
    else     { sumB[k] = rs[0]; sqB[k] = rq[0]; }
  }
}

__global__ void k_prep1(const float* sumA, const float* sqA, const float* sumB,
                        const float* sqB, const float* g1, float* scale1, float* meanAB) {
  int k = blockIdx.x * 256 + threadIdx.x;
  if (k >= 512) return;
  float mA = sumA[k] * (1.f / BSZ);
  float vA = sqA[k] * (1.f / BSZ) - mA * mA;
  float mB = sumB[k] * (1.f / NFEAT);
  float vB = sqB[k] * (1.f / NFEAT) - mB * mB;
  scale1[k]  = g1[k] * rsqrtf(vA + vB + EPSV);
  meanAB[k]  = mA + mB;
}

// B''[f,k] = (Brow - meanAB)*scale1 + beta1  (b1 cancels inside BN)
__global__ void k_makeB2(const float* __restrict__ Brow, const float* __restrict__ meanAB,
                         const float* __restrict__ scale1, const float* __restrict__ beta1,
                         float* __restrict__ B2) {
  int i = blockIdx.x * 256 + threadIdx.x;   // 32768
  int k = i & 511;
  B2[i] = (Brow[i] - meanAB[k]) * scale1[k] + beta1[k];
}

// --------------------------------------------------------- main GEMM --------
// One block per b: h1[f,k] = leaky(A[b,k]*scale1[k] + B''[f,k]) built per-lane,
// y2 = h1 @ W2 via bf16 WMMA; BN2 column sums accumulated via LDS+global atomics.
__global__ void k_main(const float* __restrict__ Amat, const float* __restrict__ scale1,
                       const float* __restrict__ B2, const unsigned* __restrict__ W2swz,
                       float* __restrict__ y2, float* sum2, float* sq2) {
  __shared__ float ArS[512];
  __shared__ float lsum[256], lsq[256];
  const int tid = threadIdx.x, lane = tid & 31, w = tid >> 5;
  const int b = blockIdx.x;
  for (int k = tid; k < 512; k += 256) ArS[k] = Amat[(size_t)b * 512 + k] * scale1[k];
  lsum[tid] = 0.f; lsq[tid] = 0.f;
  __syncthreads();
  const int mt = w >> 1, nh = w & 1;
  const int half = lane >> 4, mloc = lane & 15;
  const int frow = mt * 16 + mloc;          // feature row 0..63 held by this lane
  v8f zero = {0.f, 0.f, 0.f, 0.f, 0.f, 0.f, 0.f, 0.f};
  v8f acc[8];
#pragma unroll
  for (int i = 0; i < 8; ++i) acc[i] = zero;
  for (int kb = 0; kb < 16; ++kb) {
    FragU a;
#pragma unroll
    for (int v = 0; v < 8; ++v) {
      int k0 = kb * 32 + akidx0(v, half);
      float2 bp = *(const float2*)(B2 + (size_t)frow * 512 + k0);
      float x0 = lrelu(ArS[k0] + bp.x, 0.01f);
      float x1 = lrelu(ArS[k0 + 1] + bp.y, 0.01f);
      a.u[v] = (unsigned)f2bf(x0) | ((unsigned)f2bf(x1) << 16);
    }
#pragma unroll
    for (int i = 0; i < 8; ++i) {
      int nt = nh * 8 + i;
      const uint4* p = (const uint4*)(W2swz + (((size_t)kb * 16 + nt) * 32 + lane) * 8);
      uint4 q0 = p[0], q1 = p[1];
      FragU bb;
      bb.u[0] = q0.x; bb.u[1] = q0.y; bb.u[2] = q0.z; bb.u[3] = q0.w;
      bb.u[4] = q1.x; bb.u[5] = q1.y; bb.u[6] = q1.z; bb.u[7] = q1.w;
      acc[i] = __builtin_amdgcn_wmma_f32_16x16x32_bf16(false, a.v, false, bb.v,
                                                       (short)0, acc[i], false, false);
    }
  }
#pragma unroll
  for (int i = 0; i < 8; ++i) {
    int col = (nh * 8 + i) * 16 + mloc;
#pragma unroll
    for (int r = 0; r < 8; ++r) {
      int m = r + 8 * half;
      size_t row = (size_t)b * 64 + mt * 16 + m;
      float v = acc[i][r];
      y2[row * 256 + col] = v;
      atomicAdd(&lsum[col], v);
      atomicAdd(&lsq[col], v * v);
    }
  }
  __syncthreads();
  atomicAdd(&sum2[tid], lsum[tid]);
  atomicAdd(&sq2[tid], lsq[tid]);
}

__global__ void k_prep2(const float* sum2, const float* sq2, const float* g2,
                        const float* beta2, float* scale2, float* shift2) {
  int n = threadIdx.x;
  if (n >= 256) return;
  float m = sum2[n] * (1.f / (float)MTOT);
  float v = sq2[n] * (1.f / (float)MTOT) - m * m;
  float sc = g2[n] * rsqrtf(v + EPSV);
  scale2[n] = sc;
  shift2[n] = beta2[n] - m * sc;
}

// out[r] = b3 + sum_n leaky(y2*scale2+shift2) * W3[n] ; 32 lanes per row
__global__ void k_out(const float* __restrict__ y2, const float* __restrict__ scale2,
                      const float* __restrict__ shift2, const float* __restrict__ W3,
                      const float* __restrict__ b3, float* __restrict__ out) {
  const int tid = threadIdx.x, lane = tid & 31, g = tid >> 5;
  const size_t row = (size_t)blockIdx.x * 8 + g;
  float acc = 0.f;
#pragma unroll
  for (int j = 0; j < 8; ++j) {
    int n = lane + 32 * j;
    float v = lrelu(y2[row * 256 + n] * scale2[n] + shift2[n], 0.01f);
    acc += v * W3[n];
  }
#pragma unroll
  for (int off = 16; off > 0; off >>= 1) acc += __shfl_down(acc, off, 32);
  if (lane == 0) out[row] = acc + b3[0];
}

// ---------------------------------------------------------------------------
extern "C" void kernel_launch(void* const* d_in, const int* in_sizes, int n_in,
                              void* d_out, int out_size, void* d_ws, size_t ws_size,
                              hipStream_t stream) {
  (void)in_sizes; (void)n_in; (void)out_size; (void)ws_size;
  const float* node   = (const float*)d_in[0];
  const float* feat   = (const float*)d_in[1];
  const int*   ri     = (const int*)d_in[2];
  const float* Wg     = (const float*)d_in[3];
  const float* atts   = (const float*)d_in[4];
  const float* attd   = (const float*)d_in[5];
  const float* bgat   = (const float*)d_in[6];
  const float* W1     = (const float*)d_in[7];
  /* b1 = d_in[8] cancels in training-mode BN */
  const float* g1     = (const float*)d_in[9];
  const float* beta1  = (const float*)d_in[10];
  const float* W2     = (const float*)d_in[11];
  /* b2 = d_in[12] cancels in training-mode BN */
  const float* g2     = (const float*)d_in[13];
  const float* beta2  = (const float*)d_in[14];
  const float* W3     = (const float*)d_in[15];
  const float* b3     = (const float*)d_in[16];
  float* out = (float*)d_out;

  // -------- workspace layout --------
  float* fws    = (float*)d_ws;
  float* hfeat  = fws;                   // 32768
  float* asrc   = hfeat + 32768;         // 128
  float* adst   = asrc + 128;            // 128
  float* wedge  = adst + 128;            // 2048
  float* msg    = wedge + 2048;          // 32768  <- zero region start
  float* sum2   = msg + 32768;           // 256
  float* sq2    = sum2 + 256;            // 256    (zero region = 33280 floats)
  float* Amat   = sq2 + 256;             // 2048*512
  float* Brow   = Amat + (size_t)2048 * 512;  // 64*512
  float* sumA   = Brow + 32768;          // 512
  float* sqA    = sumA + 512;
  float* sumB   = sqA + 512;
  float* sqB    = sumB + 512;
  float* scale1 = sqB + 512;
  float* meanAB = scale1 + 512;
  float* B2     = meanAB + 512;          // 32768
  float* scale2 = B2 + 32768;            // 256
  float* shift2 = scale2 + 256;          // 256
  unsigned short* node_bf = (unsigned short*)(shift2 + 256);   // 2048*512 bf16
  unsigned short* msg_bf  = node_bf + (size_t)2048 * 512;      // 64*512 bf16
  unsigned* w1top = (unsigned*)(msg_bf + 32768);               // 131072
  unsigned* w1bot = w1top + 131072;                            // 131072
  unsigned* w2swz = w1bot + 131072;                            // 65536
  float* y2 = (float*)(w2swz + 65536);                         // 131072*256 f32

  // 0) zero accumulators (msg + BN2 sums), fresh every call
  k_zero<<<130, 256, 0, stream>>>(msg, 33280);
  // 1) GAT
  k_gat_lin<<<64, 256, 0, stream>>>(feat, Wg, hfeat);
  k_gat_att<<<1, 128, 0, stream>>>(hfeat, atts, attd, asrc, adst);
  k_gat_soft<<<1, 256, 0, stream>>>(ri, asrc, adst, wedge);
  k_gat_agg<<<EEDGE, 256, 0, stream>>>(ri, hfeat, wedge, msg);
  k_msgfinal<<<128, 256, 0, stream>>>(msg, bgat, msg_bf);
  // 2) bf16 conversions + weight swizzles
  k_cvt_bf<<<4096, 256, 0, stream>>>(node, node_bf, 2048 * 512);
  k_swz<<<512, 256, 0, stream>>>(W1, w1top, 32);                 // W1 rows 0..511
  k_swz<<<512, 256, 0, stream>>>(W1 + (size_t)512 * 512, w1bot, 32); // rows 512..1023
  k_swz<<<256, 256, 0, stream>>>(W2, w2swz, 16);
  // 3) A = node @ W1top (WMMA), Brow = msg @ W1bot (WMMA)
  k_gemm<<<dim3(32, 2), 256, 0, stream>>>(node_bf, w1top, Amat, 32);
  k_gemm<<<dim3(1, 2), 256, 0, stream>>>(msg_bf, w1bot, Brow, 32);
  // 4) closed-form BN1
  k_stats1<<<1024, 256, 0, stream>>>(Amat, Brow, sumA, sqA, sumB, sqB);
  k_prep1<<<2, 256, 0, stream>>>(sumA, sqA, sumB, sqB, g1, scale1, meanAB);
  k_makeB2<<<128, 256, 0, stream>>>(Brow, meanAB, scale1, beta1, B2);
  // 5) fused h1 generation + h1@W2 (WMMA) + BN2 stats
  k_main<<<2048, 256, 0, stream>>>(Amat, scale1, B2, w2swz, y2, sum2, sq2);
  // 6) BN2 finalize + output dot with W3
  k_prep2<<<1, 256, 0, stream>>>(sum2, sq2, g2, beta2, scale2, shift2);
  k_out<<<16384, 256, 0, stream>>>(y2, scale2, shift2, W3, b3, out);
}
